// FlatQuantPixArtAttentionBeta_56470230007882
// MI455X (gfx1250) — 8-bit matrix operand order repaired and checked by reading, compiled, not run on hardware
//
#include <hip/hip_runtime.h>
#include <hip/hip_fp16.h>
#include <stdint.h>

#define D_MODEL 1152
#define NHEADS  16
#define HD      72
#define SEQ     2048
#define BATCH   2
#define NTOK    (BATCH*SEQ)
#define LDIM    32
#define RDIM    36

typedef __attribute__((ext_vector_type(8)))  int      v8i;
typedef __attribute__((ext_vector_type(8)))  float    v8f;
typedef __attribute__((ext_vector_type(16))) _Float16 v16h;
typedef __attribute__((ext_vector_type(8)))  _Float16 v8h;

#if defined(__has_builtin)
#  if __has_builtin(__builtin_amdgcn_tensor_load_to_lds)
#    define HAVE_TDM 1
#  endif
#endif
#ifndef HAVE_TDM
#  define HAVE_TDM 0
#endif

#if HAVE_TDM
typedef __attribute__((ext_vector_type(4))) unsigned int v4u;
typedef __attribute__((ext_vector_type(4))) int          v4i;

__device__ __forceinline__ void tdm_load_2d(const void* gsrc, unsigned lds_off,
                                            unsigned tile_w, unsigned tile_h,
                                            unsigned stride_bytes) {
    unsigned long long ga = (unsigned long long)(uintptr_t)gsrc;
    v4u g0;
    g0[0] = 1u;
    g0[1] = lds_off;
    g0[2] = (unsigned)ga;
    g0[3] = (unsigned)((ga >> 32) & 0x01FFFFFFu) | (2u << 30);
    v8i g1;
    g1[0] = 0;
    g1[1] = (int)((tile_w & 0xFFFFu) << 16);
    g1[2] = (int)(((tile_w >> 16) & 0xFFFFu) |
                  ((tile_h & 0xFFFFu) << 16));
    g1[3] = (int)(((tile_h >> 16) & 0xFFFFu) |
                  ((tile_w & 0xFFFFu) << 16));
    g1[4] = (int)(tile_h & 0xFFFFu);
    g1[5] = (int)stride_bytes;
    g1[6] = 0;
    g1[7] = 0;
    v4i z4 = {0, 0, 0, 0};
#if __clang_major__ >= 23
    v8i z8 = (v8i){0, 0, 0, 0, 0, 0, 0, 0};
    __builtin_amdgcn_tensor_load_to_lds(g0, g1, z4, z4, z8, 0);
#else
    __builtin_amdgcn_tensor_load_to_lds(g0, g1, z4, z4, 0);
#endif
}
#endif

__device__ __forceinline__ float rmax16(float v) {
    #pragma unroll
    for (int m = 1; m < 16; m <<= 1) v = fmaxf(v, __shfl_xor(v, m, 32));
    return v;
}
__device__ __forceinline__ float rsum16(float v) {
    #pragma unroll
    for (int m = 1; m < 16; m <<= 1) v += __shfl_xor(v, m, 32);
    return v;
}

__global__ __launch_bounds__(128) void k_invert(const float* __restrict__ M,
                                                float* __restrict__ outInvT, int n) {
    __shared__ float A[72][145];
    int t = threadIdx.x;
    for (int idx = t; idx < n * 2 * n; idx += 128) {
        int i = idx / (2 * n), j = idx % (2 * n);
        A[i][j] = (j < n) ? M[i * n + j] : ((j - n == i) ? 1.f : 0.f);
    }
    __syncthreads();
    for (int p = 0; p < n; ++p) {
        if (t == p) {
            float inv = 1.f / A[p][p];
            for (int j = 0; j < 2 * n; ++j) A[p][j] *= inv;
        }
        __syncthreads();
        if (t < n && t != p) {
            float f = A[t][p];
            for (int j = 0; j < 2 * n; ++j) A[t][j] -= f * A[p][j];
        }
        __syncthreads();
    }
    for (int idx = t; idx < n * n; idx += 128) {
        int i = idx / n, j = idx % n;
        outInvT[i * n + j] = A[j][n + i];
    }
}

__global__ __launch_bounds__(256) void k_decomp(const float* __restrict__ in,
                                                float* __restrict__ out,
                                                const float* __restrict__ Mleft,
                                                const float* __restrict__ Mright) {
    __shared__ float Xs[D_MODEL], Ts[D_MODEL];
    __shared__ float MlS[LDIM * LDIM], MrS[RDIM * RDIM];
    int t = threadIdx.x;
    size_t row = blockIdx.x;
    for (int i = t; i < LDIM * LDIM; i += 256) MlS[i] = Mleft[i];
    for (int i = t; i < RDIM * RDIM; i += 256) MrS[i] = Mright[i];
    const float* rp = in + row * (size_t)D_MODEL;
    for (int i = t; i < D_MODEL; i += 256) Xs[i] = rp[i];
    __syncthreads();
    for (int i = t; i < D_MODEL; i += 256) {
        int L = i / RDIM, r = i % RDIM;
        float a = 0.f;
        #pragma unroll 8
        for (int l = 0; l < LDIM; ++l) a += Xs[l * RDIM + r] * MlS[l * LDIM + L];
        Ts[i] = a;
    }
    __syncthreads();
    float* op = out + row * (size_t)D_MODEL;
    for (int i = t; i < D_MODEL; i += 256) {
        int L = i / RDIM, R = i % RDIM;
        float a = 0.f;
        #pragma unroll 6
        for (int r = 0; r < RDIM; ++r) a += Ts[L * RDIM + r] * MrS[r * RDIM + R];
        op[i] = a;
    }
}

__global__ __launch_bounds__(256) void k_quant_rows(const float* __restrict__ in,
                                                    signed char* __restrict__ out,
                                                    float* __restrict__ scale,
                                                    float qmax, float qmin) {
    __shared__ float red[256];
    int t = threadIdx.x;
    size_t row = blockIdx.x;
    const float* rp = in + row * (size_t)D_MODEL;
    float m = 0.f;
    for (int i = t; i < D_MODEL; i += 256) m = fmaxf(m, fabsf(rp[i]));
    red[t] = m;
    __syncthreads();
    for (int s = 128; s > 0; s >>= 1) {
        if (t < s) red[t] = fmaxf(red[t], red[t + s]);
        __syncthreads();
    }
    float sc = fmaxf(red[0], 1e-8f) / qmax;
    if (t == 0) scale[row] = sc;
    float inv = 1.f / sc;
    signed char* op = out + row * (size_t)D_MODEL;
    for (int i = t; i < D_MODEL; i += 256) {
        float qv = fminf(fmaxf(roundf(rp[i] * inv), qmin), qmax);
        op[i] = (signed char)(int)qv;
    }
}

__global__ __launch_bounds__(256) void k_vfold(const float* __restrict__ src,
                                               float* __restrict__ dst,
                                               const float* __restrict__ Vc) {
    __shared__ float col[HD];
    int t = threadIdx.x;
    int ro = blockIdx.x;
    int h = ro / HD, Dp = ro % HD;
    if (t < HD) col[t] = Vc[t * HD + Dp];
    __syncthreads();
    const float* base = src + (size_t)h * HD * D_MODEL;
    float* op = dst + (size_t)ro * D_MODEL;
    for (int i = t; i < D_MODEL; i += 256) {
        float a = 0.f;
        #pragma unroll 8
        for (int d = 0; d < HD; ++d) a += col[d] * base[(size_t)d * D_MODEL + i];
        op[i] = a;
    }
}

__global__ void k_bvfold(const float* __restrict__ bv, const float* __restrict__ Vc,
                         float* __restrict__ bvt) {
    int h = blockIdx.x, Dp = threadIdx.x;
    if (Dp < HD) {
        float a = 0.f;
        for (int d = 0; d < HD; ++d) a += bv[h * HD + d] * Vc[d * HD + Dp];
        bvt[h * HD + Dp] = a;
    }
}

__global__ __launch_bounds__(256) void k_woprep(const float* __restrict__ Wo,
                                                const float* __restrict__ OmIT,
                                                const float* __restrict__ VcIT,
                                                float* __restrict__ out) {
    __shared__ float row[D_MODEL], tmp[D_MODEL];
    __shared__ float OmS[NHEADS * NHEADS], VcS[HD * HD];
    int t = threadIdx.x;
    size_t r = blockIdx.x;
    for (int i = t; i < NHEADS * NHEADS; i += 256) OmS[i] = OmIT[i];
    for (int i = t; i < HD * HD; i += 256) VcS[i] = VcIT[i];
    const float* rp = Wo + r * (size_t)D_MODEL;
    for (int i = t; i < D_MODEL; i += 256) row[i] = rp[i];
    __syncthreads();
    for (int i = t; i < D_MODEL; i += 256) {
        int Hp = i / HD, d = i % HD;
        float a = 0.f;
        #pragma unroll
        for (int h = 0; h < NHEADS; ++h) a += OmS[h * NHEADS + Hp] * row[h * HD + d];
        tmp[i] = a;
    }
    __syncthreads();
    float* op = out + r * (size_t)D_MODEL;
    for (int i = t; i < D_MODEL; i += 256) {
        int Hp = i / HD, Dp = i % HD;
        float a = 0.f;
        #pragma unroll 8
        for (int d = 0; d < HD; ++d) a += tmp[Hp * HD + d] * VcS[d * HD + Dp];
        op[i] = a;
    }
}

__global__ __launch_bounds__(256) void k_gemm_iu8(const signed char* __restrict__ A,
                                                  const float* __restrict__ aS,
                                                  const signed char* __restrict__ W,
                                                  const float* __restrict__ wS,
                                                  const float* __restrict__ bias,
                                                  float* __restrict__ C) {
    __shared__ __align__(16) signed char As[128][64];
    __shared__ __align__(16) signed char Bs[64][64];
    int t = threadIdx.x, w = t >> 5, l = t & 31, hi = l >> 4, l16 = l & 15;
    int wm = w & 3, wn = w >> 2;
    int m0 = blockIdx.x * 128, n0 = blockIdx.y * 64;

    v8i acc[2][2];
    #pragma unroll
    for (int i = 0; i < 2; ++i)
        #pragma unroll
        for (int j = 0; j < 2; ++j)
            acc[i][j] = (v8i){0, 0, 0, 0, 0, 0, 0, 0};

    for (int k0 = 0; k0 < D_MODEL; k0 += 64) {
        __syncthreads();
#if HAVE_TDM
        if (t < 32) {
            tdm_load_2d(A + (size_t)m0 * D_MODEL + k0,
                        (unsigned)(size_t)&As[0][0], 64u, 128u, (unsigned)D_MODEL);
        } else if (t < 64) {
            tdm_load_2d(W + (size_t)n0 * D_MODEL + k0,
                        (unsigned)(size_t)&Bs[0][0], 64u, 64u, (unsigned)D_MODEL);
        }
        if (t < 64) __builtin_amdgcn_s_wait_tensorcnt(0);
#else
        #pragma unroll
        for (int i = 0; i < 2; ++i) {
            int idx = t + i * 256;
            int rr = idx >> 2, cc = (idx & 3) * 16;
            *(int4*)&As[rr][cc] =
                *(const int4*)(A + (size_t)(m0 + rr) * D_MODEL + k0 + cc);
        }
        {
            int rr = t >> 2, cc = (t & 3) * 16;
            *(int4*)&Bs[rr][cc] =
                *(const int4*)(W + (size_t)(n0 + rr) * D_MODEL + k0 + cc);
        }
        if (k0 + 64 < D_MODEL) {
            __builtin_prefetch(A + (size_t)(m0 + (t >> 1)) * D_MODEL + k0 + 64 + (t & 1) * 32, 0, 1);
            __builtin_prefetch(W + (size_t)(n0 + (t & 63)) * D_MODEL + k0 + 64, 0, 1);
        }
#endif
        __syncthreads();

        v8i af[2], bf[2];
        int kb = hi * 16, kb2 = hi * 16;
        #pragma unroll
        for (int i = 0; i < 2; ++i) {
            const signed char* p = &As[wm * 32 + i * 16 + l16][0];
            int2 x0 = *(const int2*)(p + kb);
            int2 x1 = *(const int2*)(p + kb + 8);
            int2 x2 = *(const int2*)(p + kb + 32);
            int2 x3 = *(const int2*)(p + kb + 40);
            af[i][0] = x0.x; af[i][1] = x0.y; af[i][2] = x1.x; af[i][3] = x1.y;
            af[i][4] = x2.x; af[i][5] = x2.y; af[i][6] = x3.x; af[i][7] = x3.y;
        }
        #pragma unroll
        for (int j = 0; j < 2; ++j) {
            const signed char* p = &Bs[wn * 32 + j * 16 + l16][0];
            int4 y0 = *(const int4*)(p + kb2);
            int4 y1 = *(const int4*)(p + kb2 + 32);
            bf[j][0] = y0.x; bf[j][1] = y0.y; bf[j][2] = y0.z; bf[j][3] = y0.w;
            bf[j][4] = y1.x; bf[j][5] = y1.y; bf[j][6] = y1.z; bf[j][7] = y1.w;
        }
        #pragma unroll
        for (int i = 0; i < 2; ++i)
            #pragma unroll
            for (int j = 0; j < 2; ++j)
                acc[i][j] = __builtin_amdgcn_wmma_i32_16x16x64_iu8(
                    true, af[i], true, bf[j], acc[i][j], false, false);
    }

    #pragma unroll
    for (int i = 0; i < 2; ++i)
        #pragma unroll
        for (int j = 0; j < 2; ++j)
            #pragma unroll
            for (int r = 0; r < 8; ++r) {
                int grow = m0 + wm * 32 + i * 16 + r + 8 * hi;
                int gcol = n0 + wn * 32 + j * 16 + l16;
                C[(size_t)grow * D_MODEL + gcol] =
                    (float)acc[i][j][r] * aS[grow] * wS[gcol] + bias[gcol];
            }
}

__global__ __launch_bounds__(128) void k_rot(float* __restrict__ q, float* __restrict__ k,
                                             const float* __restrict__ v,
                                             const float* __restrict__ KcIT,
                                             const float* __restrict__ Kc,
                                             unsigned* __restrict__ amax) {
    __shared__ float KiS[HD * HD], KcS[HD * HD], rowS[HD], outS[HD], red[128];
    int t = threadIdx.x;
    size_t n = blockIdx.x;
    for (int i = t; i < HD * HD; i += 128) { KiS[i] = KcIT[i]; KcS[i] = Kc[i]; }
    __syncthreads();
    float lq = 0.f, lk = 0.f, lv = 0.f;
    for (int h = 0; h < NHEADS; ++h) {
        float* qp = q + n * D_MODEL + h * HD;
        if (t < HD) rowS[t] = qp[t];
        __syncthreads();
        if (t < HD) {
            float a = 0.f;
            #pragma unroll 8
            for (int d = 0; d < HD; ++d) a += rowS[d] * KiS[d * HD + t];
            outS[t] = a; lq = fmaxf(lq, fabsf(a));
        }
        __syncthreads();
        if (t < HD) qp[t] = outS[t];
        float* kp = k + n * D_MODEL + h * HD;
        if (t < HD) rowS[t] = kp[t];
        __syncthreads();
        if (t < HD) {
            float a = 0.f;
            #pragma unroll 8
            for (int d = 0; d < HD; ++d) a += rowS[d] * KcS[d * HD + t];
            outS[t] = a; lk = fmaxf(lk, fabsf(a));
        }
        __syncthreads();
        if (t < HD) kp[t] = outS[t];
        __syncthreads();
    }
    for (int i = t; i < D_MODEL; i += 128) lv = fmaxf(lv, fabsf(v[n * D_MODEL + i]));

    red[t] = lq; __syncthreads();
    for (int s = 64; s > 0; s >>= 1) { if (t < s) red[t] = fmaxf(red[t], red[t + s]); __syncthreads(); }
    if (t == 0) atomicMax(amax + 0, __float_as_uint(red[0]));
    __syncthreads();
    red[t] = lk; __syncthreads();
    for (int s = 64; s > 0; s >>= 1) { if (t < s) red[t] = fmaxf(red[t], red[t + s]); __syncthreads(); }
    if (t == 0) atomicMax(amax + 1, __float_as_uint(red[0]));
    __syncthreads();
    red[t] = lv; __syncthreads();
    for (int s = 64; s > 0; s >>= 1) { if (t < s) red[t] = fmaxf(red[t], red[t + s]); __syncthreads(); }
    if (t == 0) atomicMax(amax + 2, __float_as_uint(red[0]));
}

__global__ __launch_bounds__(128) void k_qattn(const float* __restrict__ q,
                                               const float* __restrict__ k,
                                               const float* __restrict__ v,
                                               const unsigned* __restrict__ amax,
                                               signed char* __restrict__ Q8,
                                               signed char* __restrict__ K8,
                                               _Float16* __restrict__ Vh) {
    int s = blockIdx.x, bh = blockIdx.y, t = threadIdx.x;
    int b = bh >> 4, h = bh & 15;
    size_t n = (size_t)b * SEQ + s;
    float qs = fmaxf(__uint_as_float(amax[0]), 1e-8f) / 127.f;
    float ks = fmaxf(__uint_as_float(amax[1]), 1e-8f) / 7.f;
    float vs = fmaxf(__uint_as_float(amax[2]), 1e-8f) / 7.f;
    size_t rowp = (size_t)bh * SEQ + s;
    float qv = 0.f, kv = 0.f;
    if (t < HD) { qv = q[n * D_MODEL + h * HD + t]; kv = k[n * D_MODEL + h * HD + t]; }
    Q8[rowp * 128 + t] = (t < HD) ?
        (signed char)(int)fminf(fmaxf(roundf(qv / qs), -128.f), 127.f) : (signed char)0;
    K8[rowp * 128 + t] = (t < HD) ?
        (signed char)(int)fminf(fmaxf(roundf(kv / ks), -8.f), 7.f) : (signed char)0;
    if (t < HD) {
        float vv = v[n * D_MODEL + h * HD + t];
        float vq = fminf(fmaxf(roundf(vv / vs), -8.f), 7.f) * vs;
        Vh[rowp * HD + t] = (_Float16)vq;
    }
}

__global__ __launch_bounds__(256) void k_flash(const signed char* __restrict__ Q8,
                                               const signed char* __restrict__ K8,
                                               const _Float16* __restrict__ Vh,
                                               const unsigned* __restrict__ amax,
                                               float* __restrict__ O) {
    __shared__ __align__(16) signed char Ks[64][128];
    __shared__ __align__(16) _Float16 Vt[80][64];
    __shared__ __align__(16) _Float16 Ps[8][16][64];
    int t = threadIdx.x, w = t >> 5, l = t & 31, hi = l >> 4, l16 = l & 15;
    int bh = blockIdx.y, b = bh >> 4, h = bh & 15;
    int q0 = blockIdx.x * 128 + w * 16;

    float qs = fmaxf(__uint_as_float(amax[0]), 1e-8f) / 127.f;
    float ks = fmaxf(__uint_as_float(amax[1]), 1e-8f) / 7.f;
    float sc = qs * ks * rsqrtf((float)HD);

    const signed char* qrow = Q8 + ((size_t)bh * SEQ + q0 + l16) * 128;
    int kb = hi * 16;
    v8i aq[2];
    #pragma unroll
    for (int c = 0; c < 2; ++c) {
        const signed char* p = qrow + c * 64 + kb;
        int2 x0 = *(const int2*)p;
        int2 x1 = *(const int2*)(p + 8);
        int2 x2 = *(const int2*)(p + 32);
        int2 x3 = *(const int2*)(p + 40);
        aq[c][0] = x0.x; aq[c][1] = x0.y; aq[c][2] = x1.x; aq[c][3] = x1.y;
        aq[c][4] = x2.x; aq[c][5] = x2.y; aq[c][6] = x3.x; aq[c][7] = x3.y;
    }

    v8f ov[5];
    #pragma unroll
    for (int nt = 0; nt < 5; ++nt) ov[nt] = (v8f){0.f, 0.f, 0.f, 0.f, 0.f, 0.f, 0.f, 0.f};
    float mrow[8], lrow[8];
    #pragma unroll
    for (int r = 0; r < 8; ++r) { mrow[r] = -1e30f; lrow[r] = 0.f; }

    for (int kt0 = 0; kt0 < SEQ; kt0 += 64) {
        __syncthreads();
#if HAVE_TDM
        if (t < 32)
            tdm_load_2d(K8 + ((size_t)bh * SEQ + kt0) * 128,
                        (unsigned)(size_t)&Ks[0][0], 128u, 64u, 128u);
#else
        #pragma unroll
        for (int i = 0; i < 2; ++i) {
            int idx = t + i * 256;
            int rr = idx >> 2, cc = (idx & 3) * 16;
            *(int4*)&Ks[rr][cc] =
                *(const int4*)(K8 + ((size_t)bh * SEQ + kt0 + rr) * 128 + cc);
        }
#endif
        for (int i = 0; i < 20; ++i) {
            int idx = t + i * 256;
            int d = idx % 80, ss = idx / 80;
            Vt[d][ss] = (d < HD) ? Vh[((size_t)bh * SEQ + kt0 + ss) * HD + d]
                                 : (_Float16)0.f;
        }
#if HAVE_TDM
        if (t < 32) __builtin_amdgcn_s_wait_tensorcnt(0);
#endif
        __syncthreads();

        float p4[4][8];
        #pragma unroll
        for (int ct = 0; ct < 4; ++ct) {
            v8i sa = (v8i){0, 0, 0, 0, 0, 0, 0, 0};
            int key = ct * 16 + l16, kb2 = hi * 16;
            #pragma unroll
            for (int c = 0; c < 2; ++c) {
                const signed char* p = &Ks[key][c * 64 + kb2];
                int4 y0 = *(const int4*)p;
                int4 y1 = *(const int4*)(p + 32);
                v8i bf;
                bf[0] = y0.x; bf[1] = y0.y; bf[2] = y0.z; bf[3] = y0.w;
                bf[4] = y1.x; bf[5] = y1.y; bf[6] = y1.z; bf[7] = y1.w;
                sa = __builtin_amdgcn_wmma_i32_16x16x64_iu8(
                    true, aq[c], true, bf, sa, false, false);
            }
            #pragma unroll
            for (int r = 0; r < 8; ++r) p4[ct][r] = (float)sa[r] * sc;
        }

        float mnew[8];
        #pragma unroll
        for (int r = 0; r < 8; ++r) {
            float m = fmaxf(fmaxf(p4[0][r], p4[1][r]), fmaxf(p4[2][r], p4[3][r]));
            m = rmax16(m);
            mnew[r] = fmaxf(mrow[r], m);
        }
        #pragma unroll
        for (int ct = 0; ct < 4; ++ct)
            #pragma unroll
            for (int r = 0; r < 8; ++r) p4[ct][r] = __expf(p4[ct][r] - mnew[r]);
        #pragma unroll
        for (int r = 0; r < 8; ++r) {
            float s4 = p4[0][r] + p4[1][r] + p4[2][r] + p4[3][r];
            s4 = rsum16(s4);
            float alpha = __expf(mrow[r] - mnew[r]);
            lrow[r] = lrow[r] * alpha + s4;
            mrow[r] = mnew[r];
            #pragma unroll
            for (int nt = 0; nt < 5; ++nt) ov[nt][r] *= alpha;
        }

        #pragma unroll
        for (int ct = 0; ct < 4; ++ct)
            #pragma unroll
            for (int r = 0; r < 8; ++r)
                Ps[w][r + 8 * hi][ct * 16 + l16] = (_Float16)p4[ct][r];
        __syncthreads();

        int kb3 = hi * 8;
        #pragma unroll
        for (int c = 0; c < 2; ++c) {
            v16h pa;
            v8h x0 = *(const v8h*)&Ps[w][l16][c * 32 + kb3];
            v8h x1 = *(const v8h*)&Ps[w][l16][c * 32 + 16 + kb3];
            #pragma unroll
            for (int i = 0; i < 8; ++i) { pa[i] = x0[i]; pa[8 + i] = x1[i]; }
            #pragma unroll
            for (int nt = 0; nt < 5; ++nt) {
                int dd = nt * 16 + l16;
                v8h y0 = *(const v8h*)&Vt[dd][c * 32 + kb3];
                v8h y1 = *(const v8h*)&Vt[dd][c * 32 + 16 + kb3];
                v16h vb;
                #pragma unroll
                for (int i = 0; i < 8; ++i) { vb[i] = y0[i]; vb[8 + i] = y1[i]; }
                ov[nt] = __builtin_amdgcn_wmma_f32_16x16x32_f16(
                    false, pa, false, vb, (short)0, ov[nt], false, false);
            }
        }
    }

    #pragma unroll
    for (int nt = 0; nt < 5; ++nt) {
        int dd = nt * 16 + l16;
        if (dd < HD) {
            #pragma unroll
            for (int r = 0; r < 8; ++r) {
                int srow = q0 + r + 8 * hi;
                size_t n = (size_t)b * SEQ + srow;
                O[n * D_MODEL + h * HD + dd] = ov[nt][r] / lrow[r];
            }
        }
    }
}

__global__ __launch_bounds__(256) void k_headmix(const float* __restrict__ o,
                                                 const float* __restrict__ Om,
                                                 float* __restrict__ o2) {
    __shared__ float row[D_MODEL], OmS[NHEADS * NHEADS];
    int t = threadIdx.x;
    size_t n = blockIdx.x;
    for (int i = t; i < NHEADS * NHEADS; i += 256) OmS[i] = Om[i];
    for (int i = t; i < D_MODEL; i += 256) row[i] = o[n * D_MODEL + i];
    __syncthreads();
    for (int i = t; i < D_MODEL; i += 256) {
        int Hp = i / HD, d = i % HD;
        float a = 0.f;
        #pragma unroll
        for (int hh = 0; hh < NHEADS; ++hh) a += OmS[hh * NHEADS + Hp] * row[hh * HD + d];
        o2[n * D_MODEL + i] = a;
    }
}

__global__ void k_zero(unsigned* p) { if (threadIdx.x < 4) p[threadIdx.x] = 0u; }

extern "C" void kernel_launch(void* const* d_in, const int* in_sizes, int n_in,
                              void* d_out, int out_size, void* d_ws, size_t ws_size,
                              hipStream_t stream) {
    (void)in_sizes; (void)n_in; (void)out_size; (void)ws_size;
    const float* hidden = (const float*)d_in[0];
    const float* Wq = (const float*)d_in[1];  const float* bq = (const float*)d_in[2];
    const float* Wk = (const float*)d_in[3];  const float* bk = (const float*)d_in[4];
    const float* Wv = (const float*)d_in[5];  const float* bv = (const float*)d_in[6];
    const float* Wo = (const float*)d_in[7];  const float* bo = (const float*)d_in[8];
    const float* Ml = (const float*)d_in[9];  const float* Mr = (const float*)d_in[10];
    const float* Kc = (const float*)d_in[11]; const float* Vc = (const float*)d_in[12];
    const float* Om = (const float*)d_in[13];

    char* wsb = (char*)d_ws;
    size_t off = 0;
    auto alloc = [&](size_t bytes) -> void* {
        void* p = (void*)(wsb + off);
        off = (off + bytes + 255) & ~(size_t)255;
        return p;
    };

    float* ht   = (float*)alloc((size_t)NTOK * D_MODEL * 4);
    float* qb   = (float*)alloc((size_t)NTOK * D_MODEL * 4);
    float* kb_  = (float*)alloc((size_t)NTOK * D_MODEL * 4);
    float* vb_  = (float*)alloc((size_t)NTOK * D_MODEL * 4);
    float* Wtmp = (float*)alloc((size_t)D_MODEL * D_MODEL * 4);
    signed char* A8 = (signed char*)alloc((size_t)NTOK * D_MODEL);
    float* aS   = (float*)alloc((size_t)NTOK * 4);
    signed char* W8q = (signed char*)alloc((size_t)D_MODEL * D_MODEL);
    signed char* W8k = (signed char*)alloc((size_t)D_MODEL * D_MODEL);
    signed char* W8v = (signed char*)alloc((size_t)D_MODEL * D_MODEL);
    signed char* W8o = (signed char*)alloc((size_t)D_MODEL * D_MODEL);
    float* wSq = (float*)alloc(D_MODEL * 4);
    float* wSk = (float*)alloc(D_MODEL * 4);
    float* wSv = (float*)alloc(D_MODEL * 4);
    float* wSo = (float*)alloc(D_MODEL * 4);
    float* MlIT = (float*)alloc(LDIM * LDIM * 4);
    float* MrIT = (float*)alloc(RDIM * RDIM * 4);
    float* KcIT = (float*)alloc(HD * HD * 4);
    float* VcIT = (float*)alloc(HD * HD * 4);
    float* OmIT = (float*)alloc(NHEADS * NHEADS * 4);
    float* bvt  = (float*)alloc(D_MODEL * 4);
    unsigned* am = (unsigned*)alloc(16);
    signed char* Q8 = (signed char*)alloc((size_t)BATCH * NHEADS * SEQ * 128);
    signed char* K8 = (signed char*)alloc((size_t)BATCH * NHEADS * SEQ * 128);
    _Float16* VhB   = (_Float16*)alloc((size_t)BATCH * NHEADS * SEQ * HD * 2);

    k_invert<<<1, 128, 0, stream>>>(Ml, MlIT, LDIM);
    k_invert<<<1, 128, 0, stream>>>(Mr, MrIT, RDIM);
    k_invert<<<1, 128, 0, stream>>>(Kc, KcIT, HD);
    k_invert<<<1, 128, 0, stream>>>(Vc, VcIT, HD);
    k_invert<<<1, 128, 0, stream>>>(Om, OmIT, NHEADS);

    k_decomp<<<NTOK, 256, 0, stream>>>(hidden, ht, Ml, Mr);
    k_quant_rows<<<NTOK, 256, 0, stream>>>(ht, A8, aS, 127.f, -128.f);

    k_decomp<<<D_MODEL, 256, 0, stream>>>(Wq, Wtmp, MlIT, MrIT);
    k_quant_rows<<<D_MODEL, 256, 0, stream>>>(Wtmp, W8q, wSq, 7.f, -8.f);
    k_decomp<<<D_MODEL, 256, 0, stream>>>(Wk, Wtmp, MlIT, MrIT);
    k_quant_rows<<<D_MODEL, 256, 0, stream>>>(Wtmp, W8k, wSk, 7.f, -8.f);
    k_decomp<<<D_MODEL, 256, 0, stream>>>(Wv, Wtmp, MlIT, MrIT);
    k_vfold<<<D_MODEL, 256, 0, stream>>>(Wtmp, ht, Vc);
    k_quant_rows<<<D_MODEL, 256, 0, stream>>>(ht, W8v, wSv, 7.f, -8.f);
    k_bvfold<<<NHEADS, HD, 0, stream>>>(bv, Vc, bvt);
    k_woprep<<<D_MODEL, 256, 0, stream>>>(Wo, OmIT, VcIT, Wtmp);
    k_quant_rows<<<D_MODEL, 256, 0, stream>>>(Wtmp, W8o, wSo, 7.f, -8.f);

    dim3 gg(NTOK / 128, D_MODEL / 64);
    k_gemm_iu8<<<gg, 256, 0, stream>>>(A8, aS, W8q, wSq, bq, qb);
    k_gemm_iu8<<<gg, 256, 0, stream>>>(A8, aS, W8k, wSk, bk, kb_);
    k_gemm_iu8<<<gg, 256, 0, stream>>>(A8, aS, W8v, wSv, bvt, vb_);

    k_zero<<<1, 32, 0, stream>>>(am);
    k_rot<<<NTOK, 128, 0, stream>>>(qb, kb_, vb_, KcIT, Kc, am);
    dim3 gq(SEQ, BATCH * NHEADS);
    k_qattn<<<gq, 128, 0, stream>>>(qb, kb_, vb_, am, Q8, K8, VhB);

    dim3 gf(SEQ / 128, BATCH * NHEADS);
    k_flash<<<gf, 256, 0, stream>>>(Q8, K8, VhB, am, qb);

    k_headmix<<<NTOK, 256, 0, stream>>>(qb, Om, ht);
    k_quant_rows<<<NTOK, 256, 0, stream>>>(ht, A8, aS, 127.f, -128.f);
    k_gemm_iu8<<<gg, 256, 0, stream>>>(A8, aS, W8o, wSo, bo, (float*)d_out);
}
